// LlamaAttention_63857573757143
// MI455X (gfx1250) — compile-verified, matrix operand order checked by reading, not run on hardware
//
#include <hip/hip_runtime.h>
#include <hip/hip_bf16.h>
#include <math.h>

// ---------------------------------------------------------------------------
// LlamaAttention forward for MI455X (gfx1250, wave32, WMMA bf16->f32)
//   B=2, S=2048, DIM=2048, H=16, hd=128
// ---------------------------------------------------------------------------

#define BATCH   2
#define SEQ     2048
#define DIM     2048
#define NHEADS  16
#define HDIM    128
#define BS      (BATCH * SEQ)          // 4096 rows

typedef __attribute__((ext_vector_type(16))) __bf16 v16bf;
typedef __attribute__((ext_vector_type(8)))  __bf16 v8bf;
typedef __attribute__((ext_vector_type(8)))  float  v8f;

// ---- WMMA helper -----------------------------------------------------------
__device__ __forceinline__ v8f wmma_bf16(v16bf a, v16bf b, v8f c) {
  return __builtin_amdgcn_wmma_f32_16x16x32_bf16(
      /*neg_a=*/false, a, /*neg_b=*/false, b,
      /*c_mod=*/(short)0, c, /*reuse_a=*/false, /*reuse_b=*/false);
}

// 16-bit 16x32 A-fragment / 32x16 B-fragment load.
// Per CDNA5 ISA layout, each lane needs two contiguous 8-element (16B) runs:
//   lanes 0-15 : K = 0..7  and 16..23
//   lanes 16-31: K = 8..15 and 24..31
// p points at [row_for_this_lane][k0]; khalf = lane>>4.
__device__ __forceinline__ v16bf frag_ld(const __bf16* p, int khalf) {
  v8bf lo = *(const v8bf*)(p + khalf * 8);
  v8bf hi = *(const v8bf*)(p + 16 + khalf * 8);
  v16bf r;
#pragma unroll
  for (int i = 0; i < 8; ++i) { r[i] = lo[i]; r[i + 8] = hi[i]; }
  return r;
}

// ---- fp32 -> bf16 bulk convert --------------------------------------------
__global__ __launch_bounds__(256) void cvt_bf16_kernel(
    const float* __restrict__ x, __bf16* __restrict__ y, int n) {
  int i = blockIdx.x * blockDim.x + threadIdx.x;
  if (i < n) y[i] = (__bf16)x[i];
}

// ---- GEMM: C[M,N] = A[M,K] (bf16) @ W[N,K]^T (bf16), fp32 out -------------
// Block = 128 threads = 4 waves (2x2), each wave computes a 64x64 tile as
// 4x4 WMMA accumulators. Operands streamed from global (fits MI455X 192MB L2).
__global__ __launch_bounds__(128) void gemm_xwt_kernel(
    const __bf16* __restrict__ A, const __bf16* __restrict__ W,
    float* __restrict__ C, int M, int N, int K) {
  const int lane  = threadIdx.x & 31;
  const int wave  = threadIdx.x >> 5;
  const int khalf = lane >> 4;
  const int ln    = lane & 15;
  const int m0 = blockIdx.y * 128 + (wave >> 1) * 64;
  const int n0 = blockIdx.x * 128 + (wave & 1) * 64;

  v8f acc[4][4] = {};

  const __bf16* aBase = A + (size_t)(m0 + ln) * K;
  const __bf16* wBase = W + (size_t)(n0 + ln) * K;

  for (int k0 = 0; k0 < K; k0 += 32) {
    __builtin_prefetch(aBase + k0 + 512, 0, 1);
    __builtin_prefetch(wBase + k0 + 512, 0, 1);
    v16bf af[4], bf[4];
#pragma unroll
    for (int t = 0; t < 4; ++t)
      af[t] = frag_ld(aBase + (size_t)t * 16 * K + k0, khalf);
#pragma unroll
    for (int t = 0; t < 4; ++t)
      bf[t] = frag_ld(wBase + (size_t)t * 16 * K + k0, khalf);
#pragma unroll
    for (int i = 0; i < 4; ++i)
#pragma unroll
      for (int j = 0; j < 4; ++j)
        acc[i][j] = wmma_bf16(af[i], bf[j], acc[i][j]);
  }

  // C fragment layout: lane holds col n = ln, rows m = v + 8*khalf.
#pragma unroll
  for (int i = 0; i < 4; ++i)
#pragma unroll
    for (int j = 0; j < 4; ++j) {
      const int n = n0 + j * 16 + ln;
#pragma unroll
      for (int v = 0; v < 8; ++v) {
        const int m = m0 + i * 16 + khalf * 8 + v;
        C[(size_t)m * N + n] = acc[i][j][v];
      }
    }
}

// ---- RoPE + repack: fp32 [B,S,D] -> bf16 [B,H,S,hd] -----------------------
__global__ __launch_bounds__(256) void rope_pack_kernel(
    const float* __restrict__ X, const int* __restrict__ pos,
    __bf16* __restrict__ Y) {
  // one thread per rotation pair: idx over B*S*H*(hd/2) = 2^22 threads
  int idx = blockIdx.x * blockDim.x + threadIdx.x;
  int d = idx & 63;
  int h = (idx >> 6) & (NHEADS - 1);
  int s = (idx >> 10) & (SEQ - 1);
  int b = idx >> 21;
  if (b >= BATCH) return;

  const float* row = X + (size_t)(b * SEQ + s) * DIM + h * HDIM;
  float x1 = row[d];
  float x2 = row[d + 64];
  float p  = (float)pos[b * SEQ + s];
  float inv = powf(10000.0f, -(float)d * (1.0f / 64.0f));
  float ang = p * inv;
  float c = cosf(ang);
  float sn = sinf(ang);
  __bf16* out = Y + (size_t)((b * NHEADS + h) * SEQ + s) * HDIM;
  out[d]      = (__bf16)(x1 * c - x2 * sn);
  out[d + 64] = (__bf16)(x2 * c + x1 * sn);
}

// ---- V repack transposed: fp32 [B,S,D] -> bf16 [B,H,hd,S] -----------------
__global__ __launch_bounds__(256) void v_trans_kernel(
    const float* __restrict__ X, __bf16* __restrict__ Y) {
  int idx = blockIdx.x * blockDim.x + threadIdx.x;  // B*S*D = 2^23
  int d = idx & (HDIM - 1);
  int h = (idx >> 7) & (NHEADS - 1);
  int s = (idx >> 11) & (SEQ - 1);
  int b = idx >> 22;
  if (b >= BATCH) return;
  float x = X[(size_t)(b * SEQ + s) * DIM + h * HDIM + d];
  Y[((size_t)(b * NHEADS + h) * HDIM + d) * SEQ + s] = (__bf16)x;
}

// ---- Flash attention: Qb/Kb [B,H,S,hd] bf16, Vt [B,H,hd,S] bf16 -----------
// Grid: (B*H, S/64); 128 threads = 4 waves; each wave owns 16 query rows.
// Online softmax; QK^T and PV both via v_wmma_f32_16x16x32_bf16.
__global__ __launch_bounds__(128) void attn_kernel(
    const __bf16* __restrict__ Qb, const __bf16* __restrict__ Kb,
    const __bf16* __restrict__ Vt, const float* __restrict__ mask,
    __bf16* __restrict__ Ob) {
  const int lane  = threadIdx.x & 31;
  const int wave  = threadIdx.x >> 5;
  const int khalf = lane >> 4;
  const int ln    = lane & 15;
  const int bh = blockIdx.x;                 // b*H + h
  const int b  = bh >> 4;
  const int h  = bh & (NHEADS - 1);
  const int q0 = blockIdx.y * 64 + wave * 16;

  // per-wave 16x32 bf16 P staging patch (C-layout -> A-layout transpose)
  __shared__ __bf16 lds_p[4][16 * 32 + 8];   // stride keeps 16B alignment

  const __bf16* qhead = Qb + (size_t)bh * SEQ * HDIM;
  const __bf16* khead = Kb + (size_t)bh * SEQ * HDIM;
  const __bf16* vth   = Vt + (size_t)bh * HDIM * SEQ;

  // Q fragments: 16 rows x 128 dims = 4 A-fragments, kept in VGPRs
  v16bf qf[4];
#pragma unroll
  for (int t = 0; t < 4; ++t)
    qf[t] = frag_ld(qhead + (size_t)(q0 + ln) * HDIM + t * 32, khalf);

  v8f o[8] = {};                 // 16x128 fp32 output accumulator (8 tiles)
  float mrow[8], lsum[8];
#pragma unroll
  for (int v = 0; v < 8; ++v) { mrow[v] = -1e30f; lsum[v] = 0.0f; }

  const float scale = 0.08838834764831845f;  // 1/sqrt(128)
  const float L2E   = 1.4426950408889634f;

  for (int kb = 0; kb < SEQ; kb += 32) {
    // scores: 16 queries x 32 keys (two 16x16 tiles), K-dim = hd = 128
    v8f s0 = {}, s1 = {};
#pragma unroll
    for (int t = 0; t < 4; ++t) {
      v16bf kf = frag_ld(khead + (size_t)(kb + ln) * HDIM + t * 32, khalf);
      s0 = wmma_bf16(qf[t], kf, s0);
    }
#pragma unroll
    for (int t = 0; t < 4; ++t) {
      v16bf kf = frag_ld(khead + (size_t)(kb + 16 + ln) * HDIM + t * 32, khalf);
      s1 = wmma_bf16(qf[t], kf, s1);
    }

    // online softmax update, rows m = khalf*8 + v live across lanes ln=0..15
#pragma unroll
    for (int v = 0; v < 8; ++v) {
      const int m = khalf * 8 + v;
      float a = s0[v] * scale + mask[(size_t)(q0 + m) * SEQ + kb + ln];
      float c = s1[v] * scale + mask[(size_t)(q0 + m) * SEQ + kb + 16 + ln];
      float mx = fmaxf(a, c);
      mx = fmaxf(mx, __shfl_xor(mx, 1, 32));
      mx = fmaxf(mx, __shfl_xor(mx, 2, 32));
      mx = fmaxf(mx, __shfl_xor(mx, 4, 32));
      mx = fmaxf(mx, __shfl_xor(mx, 8, 32));
      float nm   = fmaxf(mrow[v], mx);
      float corr = exp2f((mrow[v] - nm) * L2E);
      float p0   = exp2f((a - nm) * L2E);
      float p1   = exp2f((c - nm) * L2E);
      float rs = p0 + p1;
      rs += __shfl_xor(rs, 1, 32);
      rs += __shfl_xor(rs, 2, 32);
      rs += __shfl_xor(rs, 4, 32);
      rs += __shfl_xor(rs, 8, 32);
      lsum[v] = lsum[v] * corr + rs;
      mrow[v] = nm;
#pragma unroll
      for (int t = 0; t < 8; ++t) o[t][v] = o[t][v] * corr;
      lds_p[wave][m * 32 + ln]      = (__bf16)p0;
      lds_p[wave][m * 32 + 16 + ln] = (__bf16)p1;
    }

    // P (16x32 keys) as A-fragment from LDS (same-wave DS ops are in-order)
    v16bf pf = frag_ld(&lds_p[wave][ln * 32], khalf);

    // O += P @ V : 8 d-tiles, V^T rows contiguous over keys
#pragma unroll
    for (int dt = 0; dt < 8; ++dt) {
      v16bf vf = frag_ld(vth + (size_t)(dt * 16 + ln) * SEQ + kb, khalf);
      o[dt] = wmma_bf16(pf, vf, o[dt]);
    }
  }

  // normalize and emit bf16 [B,S,D] for the output projection
#pragma unroll
  for (int dt = 0; dt < 8; ++dt)
#pragma unroll
    for (int v = 0; v < 8; ++v) {
      const int m = khalf * 8 + v;
      float val = o[dt][v] / lsum[v];
      Ob[(size_t)(b * SEQ + q0 + m) * DIM + h * HDIM + dt * 16 + ln] =
          (__bf16)val;
    }
}

// ---------------------------------------------------------------------------
extern "C" void kernel_launch(void* const* d_in, const int* in_sizes, int n_in,
                              void* d_out, int out_size, void* d_ws,
                              size_t ws_size, hipStream_t stream) {
  (void)in_sizes; (void)n_in; (void)out_size; (void)ws_size;

  const float* hs   = (const float*)d_in[0];   // [B,S,D]
  const float* mask = (const float*)d_in[1];   // [1,1,S,S]
  const int*   pos  = (const int*)d_in[2];     // [B,S]
  const float* wq   = (const float*)d_in[3];   // [D,D]
  const float* wk   = (const float*)d_in[4];
  const float* wv   = (const float*)d_in[5];
  const float* wo   = (const float*)d_in[6];
  float* out = (float*)d_out;                  // [B,S,D] fp32

  char* ws = (char*)d_ws;
  const size_t MB = 1024 * 1024;
  __bf16* XB  = (__bf16*)(ws + 0);             // x bf16        16 MB
  __bf16* WQB = (__bf16*)(ws + 16 * MB);       // weights bf16   8 MB each
  __bf16* WKB = (__bf16*)(ws + 24 * MB);
  __bf16* WVB = (__bf16*)(ws + 32 * MB);
  __bf16* WOB = (__bf16*)(ws + 40 * MB);
  float*  PF  = (float*)(ws + 48 * MB);        // fp32 proj out 32 MB (reused)
  __bf16* QB  = (__bf16*)(ws + 80 * MB);       // [B,H,S,hd]    16 MB
  __bf16* KB  = (__bf16*)(ws + 96 * MB);
  __bf16* VT  = (__bf16*)(ws + 112 * MB);      // [B,H,hd,S]    16 MB
  __bf16* AT  = (__bf16*)(ws + 128 * MB);      // attn out bf16 16 MB

  const int nX = BS * DIM;                     // 8,388,608
  const int nW = DIM * DIM;                    // 4,194,304

  // 1) fp32 -> bf16 operand conversion
  cvt_bf16_kernel<<<nX / 256, 256, 0, stream>>>(hs, XB, nX);
  cvt_bf16_kernel<<<nW / 256, 256, 0, stream>>>(wq, WQB, nW);
  cvt_bf16_kernel<<<nW / 256, 256, 0, stream>>>(wk, WKB, nW);
  cvt_bf16_kernel<<<nW / 256, 256, 0, stream>>>(wv, WVB, nW);
  cvt_bf16_kernel<<<nW / 256, 256, 0, stream>>>(wo, WOB, nW);

  dim3 ggrid(DIM / 128, BS / 128);             // (16, 32)
  dim3 gblk(128);

  // 2) Q projection + RoPE
  gemm_xwt_kernel<<<ggrid, gblk, 0, stream>>>(XB, WQB, PF, BS, DIM, DIM);
  rope_pack_kernel<<<(BS * NHEADS * 64) / 256, 256, 0, stream>>>(PF, pos, QB);

  // 3) K projection + RoPE
  gemm_xwt_kernel<<<ggrid, gblk, 0, stream>>>(XB, WKB, PF, BS, DIM, DIM);
  rope_pack_kernel<<<(BS * NHEADS * 64) / 256, 256, 0, stream>>>(PF, pos, KB);

  // 4) V projection + transposed repack
  gemm_xwt_kernel<<<ggrid, gblk, 0, stream>>>(XB, WVB, PF, BS, DIM, DIM);
  v_trans_kernel<<<nX / 256, 256, 0, stream>>>(PF, VT);

  // 5) flash attention
  dim3 agrid(BATCH * NHEADS, SEQ / 64);        // (32, 32)
  attn_kernel<<<agrid, dim3(128), 0, stream>>>(QB, KB, VT, mask, AT);

  // 6) output projection -> fp32 d_out
  gemm_xwt_kernel<<<ggrid, gblk, 0, stream>>>(AT, WOB, out, BS, DIM, DIM);
}
